// PaiNNMessage_11089605558373
// MI455X (gfx1250) — compile-verified, matrix operand order checked by reading, not run on hardware
//
#include <hip/hip_runtime.h>
#include <hip/hip_bf16.h>

#define F_DIM   128
#define E_DIM   20
#define CUT     5.0f
#define PI_F    3.14159265358979f

typedef _Float16 half16  __attribute__((ext_vector_type(16)));
typedef _Float16 half8   __attribute__((ext_vector_type(8)));
typedef _Float16 half4v  __attribute__((ext_vector_type(4)));
typedef float    floatx8 __attribute__((ext_vector_type(8)));

// workspace layout (bytes)
#define WS_W1H   0u          // 128*128 f16  (32 KB)
#define WS_W2H   32768u      // 384*128 f16  (96 KB)
#define WS_WWH   131072u     // 384*32  f16  (24 KB, K padded 20->32 with zeros)
#define WS_PHII  155648u     // n_nodes*128*4 f16, interleaved [node][f][seg0..2,pad] (51.2 MB)

// Pack a v16h WMMA operand fragment from a 32-half K-chunk of one row/col.
// 16-bit A/B layout: lane<16 holds K {0..7,16..23}, lane>=16 holds K {8..15,24..31}.
__device__ __forceinline__ half16 frag_from(const _Float16* __restrict__ row, int hi) {
    const _Float16* p = row + hi * 8;
    half8 lo = *(const half8*)(p);
    half8 hh = *(const half8*)(p + 16);
    half16 r;
#pragma unroll
    for (int i = 0; i < 8; ++i) { r[i] = lo[i]; r[i + 8] = hh[i]; }
    return r;
}

__device__ __forceinline__ floatx8 wmma_f16(half16 a, half16 b, floatx8 c) {
    return __builtin_amdgcn_wmma_f32_16x16x32_f16(false, a, false, b, (short)0, c,
                                                  false, false);
}

// ---------------------------------------------------------------- prep
__global__ void __launch_bounds__(256) prep_weights(
        const float* __restrict__ w1, const float* __restrict__ w2,
        const float* __restrict__ ww,
        _Float16* __restrict__ w1h, _Float16* __restrict__ w2h,
        _Float16* __restrict__ wwh) {
    int i = blockIdx.x * 256 + threadIdx.x;
    if (i < 128 * 128) w1h[i] = (_Float16)w1[i];
    if (i < 384 * 128) w2h[i] = (_Float16)w2[i];
    if (i < 384 * 32) {
        int n = i >> 5, k = i & 31;
        wwh[i] = (k < E_DIM) ? (_Float16)ww[n * E_DIM + k] : (_Float16)0.0f;
    }
}

// ---------------------------------------------------------------- phi MLP
// 16 nodes per block, 128 threads (4 waves). phi = silu(state@W1^T+b1)@W2^T+b2
// Output stored interleaved: phi_i[(node*128 + f)*4 + seg], seg 0..2, slot 3 not written.
__global__ void __launch_bounds__(128) phi_kernel(
        const float* __restrict__ state,
        const float* __restrict__ b1, const float* __restrict__ b2,
        const _Float16* __restrict__ w1h, const _Float16* __restrict__ w2h,
        _Float16* __restrict__ phi_i) {
    __shared__ __align__(16) _Float16 Sh[16 * 128];
    __shared__ __align__(16) _Float16 Hh[16 * 128];
    const int base = blockIdx.x * 16;
    const int t = threadIdx.x;

    // stage + convert 16 state rows to f16
#pragma unroll
    for (int i = 0; i < 16; ++i) {
        int idx = t + i * 128;
        Sh[idx] = (_Float16)state[(size_t)base * 128 + idx];
    }
    __syncthreads();

    const int w = t >> 5, lane = t & 31;
    const int col = lane & 15, hi = lane >> 4;

    // layer 1: 8 N-tiles of 16, wave w handles ct = w, w+4
#pragma unroll
    for (int j = 0; j < 2; ++j) {
        int ct = w + 4 * j;
        int n = ct * 16 + col;
        floatx8 acc = {};
#pragma unroll
        for (int kk = 0; kk < 4; ++kk) {
            half16 a = frag_from(&Sh[col * 128 + kk * 32], hi);
            half16 b = frag_from(&w1h[n * 128 + kk * 32], hi);
            acc = wmma_f16(a, b, acc);
        }
        float bias = b1[n];
#pragma unroll
        for (int r = 0; r < 8; ++r) {
            float x = acc[r] + bias;
            float s = x / (1.0f + __expf(-x));          // silu
            Hh[(r + hi * 8) * 128 + n] = (_Float16)s;
        }
    }
    __syncthreads();

    // layer 2: 24 N-tiles, wave w handles ct = w + 4*j
#pragma unroll
    for (int j = 0; j < 6; ++j) {
        int ct = w + 4 * j;
        int nfull = ct * 16 + col;                 // 0..383
        int seg = ct >> 3;                         // 0..2
        int f   = (ct & 7) * 16 + col;             // channel 0..127
        floatx8 acc = {};
#pragma unroll
        for (int kk = 0; kk < 4; ++kk) {
            half16 a = frag_from(&Hh[col * 128 + kk * 32], hi);
            half16 b = frag_from(&w2h[nfull * 128 + kk * 32], hi);
            acc = wmma_f16(a, b, acc);
        }
        float bias = b2[nfull];
#pragma unroll
        for (int r = 0; r < 8; ++r) {
            int m = r + hi * 8;
            phi_i[((size_t)(base + m) * 128 + f) * 4 + seg] = (_Float16)(acc[r] + bias);
        }
    }
}

// ---------------------------------------------------------------- init out
__global__ void __launch_bounds__(256) init_out(
        const float4* __restrict__ s4, const float4* __restrict__ v4,
        float4* __restrict__ o4, long long ns4, long long tot4) {
    long long i = (long long)blockIdx.x * 256 + threadIdx.x;
    if (i >= tot4) return;
    o4[i] = (i < ns4) ? s4[i] : v4[i - ns4];
}

// ---------------------------------------------------------------- edges
// 16 edges per block, 128 threads (4 waves).
__global__ void __launch_bounds__(128) edge_kernel(
        const int*   __restrict__ edge,       // (2, E): row0 = dst, row1 = src
        const float* __restrict__ r_ij,       // (E, 3)
        const float* __restrict__ norm_r,     // (E)
        const float* __restrict__ W_b,        // (384)
        const float* __restrict__ state_vec,  // (N, 128, 3)
        const _Float16* __restrict__ wwh,     // (384, 32) f16
        const half4v* __restrict__ phi4,      // (N, 128) x {seg0,seg1,seg2,pad} f16
        float* __restrict__ out_s,            // (N, 128)
        float* __restrict__ out_v,            // (N, 128, 3)
        int n_edges) {
    __shared__ __align__(16) _Float16 Ah[16 * 32];  // rbf tile, K padded to 32
    __shared__ float s_cc[16];
    __shared__ float s_nr[16][3];
    __shared__ int   s_src[16];
    __shared__ int   s_dst[16];

    const int t = threadIdx.x;
    const int base = blockIdx.x * 16;

    if (t < 16) {
        int e = base + t;
        float nr = norm_r[e];
        s_cc[t]  = (nr < CUT) ? 0.5f * (__cosf(PI_F * nr * (1.0f / CUT)) + 1.0f) : 0.0f;
        s_dst[t] = edge[e];
        s_src[t] = edge[n_edges + e];
        float inv = 1.0f / nr;
        s_nr[t][0] = r_ij[3 * e + 0] * inv;
        s_nr[t][1] = r_ij[3 * e + 1] * inv;
        s_nr[t][2] = r_ij[3 * e + 2] * inv;
    }
    // rbf A-tile: 16 edges x 32 slots (20 real + 12 zero pad)
#pragma unroll
    for (int i = 0; i < 4; ++i) {
        int idx = t + i * 128;          // 0..511
        int e = idx >> 5, d = idx & 31;
        float nr = norm_r[base + e];
        float v = 0.0f;
        if (d < E_DIM) v = __sinf(nr * (float)(d + 1) * (PI_F / CUT)) / nr;
        Ah[idx] = (_Float16)v;
    }
    __syncthreads();

    const int w = t >> 5, lane = t & 31;
    const int col = lane & 15, hi = lane >> 4;

    half16 a0 = frag_from(&Ah[col * 32], hi);   // shared A operand for all tiles

#pragma unroll
    for (int j = 0; j < 2; ++j) {
        int ct = w + 4 * j;
        int n = ct * 16 + col;                  // channel f in 0..127
        // same channels across the three 128-wide segments of Wf
        half16 bg = frag_from(&wwh[(n       ) * 32], hi);
        half16 be = frag_from(&wwh[(n + 128 ) * 32], hi);
        half16 bs = frag_from(&wwh[(n + 256 ) * 32], hi);
        floatx8 c0 = {}, c1 = {}, c2 = {};
        c0 = wmma_f16(a0, bg, c0);
        c1 = wmma_f16(a0, be, c1);
        c2 = wmma_f16(a0, bs, c2);
        float wb0 = W_b[n], wb1 = W_b[128 + n], wb2 = W_b[256 + n];
#pragma unroll
        for (int r = 0; r < 8; ++r) {
            int m = r + hi * 8;                 // edge row in tile
            float cc = s_cc[m];
            int s = s_src[m], d = s_dst[m];
            half4v p = phi4[(size_t)s * 128 + n];          // one b64 gather: 3 segs
            float gsv = (c0[r] + wb0) * cc * (float)p[0];  // gate_state_vec
            float gev = (c1[r] + wb1) * cc * (float)p[1];  // gate_edge_vec
            float sm  = (c2[r] + wb2) * cc * (float)p[2];  // scalar_message
            atomicAdd(&out_s[(size_t)d * 128 + n], sm);
            const float3* sv3 = (const float3*)(state_vec + ((size_t)s * 128 + n) * 3);
            float3 sv = *sv3;                              // one b96 gather
            size_t ovb = ((size_t)d * 128 + n) * 3;
            atomicAdd(&out_v[ovb + 0], sv.x * gsv + s_nr[m][0] * gev);
            atomicAdd(&out_v[ovb + 1], sv.y * gsv + s_nr[m][1] * gev);
            atomicAdd(&out_v[ovb + 2], sv.z * gsv + s_nr[m][2] * gev);
        }
    }
}

// ---------------------------------------------------------------- launch
extern "C" void kernel_launch(void* const* d_in, const int* in_sizes, int n_in,
                              void* d_out, int out_size, void* d_ws, size_t ws_size,
                              hipStream_t stream) {
    const float* state   = (const float*)d_in[0];
    const float* statev  = (const float*)d_in[1];
    const int*   edge    = (const int*)  d_in[2];
    const float* r_ij    = (const float*)d_in[3];
    const float* norm_r  = (const float*)d_in[4];
    const float* phi_w1  = (const float*)d_in[5];
    const float* phi_b1  = (const float*)d_in[6];
    const float* phi_w2  = (const float*)d_in[7];
    const float* phi_b2  = (const float*)d_in[8];
    const float* W_w     = (const float*)d_in[9];
    const float* W_b     = (const float*)d_in[10];

    const int n_nodes = in_sizes[0] / F_DIM;     // 50000
    const int n_edges = in_sizes[4];             // 800000

    char* ws = (char*)d_ws;
    _Float16* w1h   = (_Float16*)(ws + WS_W1H);
    _Float16* w2h   = (_Float16*)(ws + WS_W2H);
    _Float16* wwh   = (_Float16*)(ws + WS_WWH);
    _Float16* phi_i = (_Float16*)(ws + WS_PHII);

    float* out_s = (float*)d_out;
    float* out_v = out_s + (size_t)n_nodes * F_DIM;

    // 1) weights -> f16
    prep_weights<<<(384 * 128 + 255) / 256, 256, 0, stream>>>(
        phi_w1, phi_w2, W_w, w1h, w2h, wwh);

    // 2) node MLP (phi) -> interleaved f16 workspace
    phi_kernel<<<(n_nodes + 15) / 16, 128, 0, stream>>>(
        state, phi_b1, phi_b2, w1h, w2h, phi_i);

    // 3) out = [state, state_vec]
    long long ns4  = (long long)n_nodes * F_DIM / 4;        // 1.6e6 float4
    long long tot4 = (long long)n_nodes * F_DIM;            // 6.4e6 float4 total
    init_out<<<(int)((tot4 + 255) / 256), 256, 0, stream>>>(
        (const float4*)state, (const float4*)statev, (float4*)d_out, ns4, tot4);

    // 4) edge messages + atomic scatter
    edge_kernel<<<(n_edges + 15) / 16, 128, 0, stream>>>(
        edge, r_ij, norm_r, W_b, statev, wwh, (const half4v*)phi_i,
        out_s, out_v, n_edges);
}
